// LSH_34780645163546
// MI455X (gfx1250) — matrix operand order repaired and checked by reading, not compiled, not run on hardware
//
#include <hip/hip_runtime.h>

// Problem dims
#define BN   8
#define CCH  64
#define HH   128
#define WW   128
#define LW   126          // H - 3 + 1
#define KTOT 576          // 64 * 9
#define EPSF 1e-12f

typedef float  v8f    __attribute__((ext_vector_type(8)));
typedef __bf16 bf16x8  __attribute__((ext_vector_type(8)));
typedef __bf16 bf16x16 __attribute__((ext_vector_type(16)));
typedef unsigned int u32x4 __attribute__((ext_vector_type(4)));
typedef int          i32x4 __attribute__((ext_vector_type(4)));
typedef int          i32x8 __attribute__((ext_vector_type(8)));

union Frag16 { bf16x8 h[2]; bf16x16 v; };

#if defined(__has_builtin)
#if __has_builtin(__builtin_amdgcn_tensor_load_to_lds) && \
    __has_builtin(__builtin_amdgcn_s_wait_tensorcnt)
#define USE_TDM 1
#endif
#endif

// ---------------------------------------------------------------------------
// Stage 1: sq[b,y,x] = sum_c x[b,c,y,x]^2           (8*128*128 threads)
// ---------------------------------------------------------------------------
__global__ __launch_bounds__(256) void k_sq(const float* __restrict__ x,
                                            float* __restrict__ sq) {
  int idx = blockIdx.x * 256 + threadIdx.x;      // [0, 131072)
  int b   = idx >> 14;
  int yx  = idx & 16383;
  const float* p = x + (size_t)b * CCH * HH * WW + yx;
  float s = 0.f;
#pragma unroll 8
  for (int c = 0; c < CCH; ++c) {
    float v = p[(size_t)c * HH * WW];
    s = fmaf(v, v, s);
  }
  sq[idx] = s;
}

// ---------------------------------------------------------------------------
// Stage 2: inv[b,p,q] = 1 / max(sqrt(3x3 boxsum(sq)), eps)   (8*126*126)
// ---------------------------------------------------------------------------
__global__ __launch_bounds__(256) void k_inv(const float* __restrict__ sq,
                                             float* __restrict__ inv) {
  int idx = blockIdx.x * 256 + threadIdx.x;
  if (idx >= BN * LW * LW) return;
  int b  = idx / (LW * LW);
  int r  = idx % (LW * LW);
  int ph = r / LW, pw = r % LW;
  const float* p = sq + b * HH * WW + ph * WW + pw;
  float s = 0.f;
#pragma unroll
  for (int i = 0; i < 3; ++i) {
#pragma unroll
    for (int j = 0; j < 3; ++j) s += p[i * WW + j];
  }
  inv[idx] = 1.0f / fmaxf(sqrtf(s), EPSF);
}

// ---------------------------------------------------------------------------
// Stage 3: S[b,y,x] = sum_{i,j} inv[b, y-i, x-j] (clipped)   (8*128*128)
// ---------------------------------------------------------------------------
__global__ __launch_bounds__(256) void k_scale(const float* __restrict__ inv,
                                               float* __restrict__ S) {
  int idx = blockIdx.x * 256 + threadIdx.x;      // [0, 131072)
  int b  = idx >> 14;
  int y  = (idx >> 7) & 127;
  int xx = idx & 127;
  float s = 0.f;
#pragma unroll
  for (int i = 0; i < 3; ++i) {
    int ph = y - i;
    if ((unsigned)ph < LW) {
#pragma unroll
      for (int j = 0; j < 3; ++j) {
        int pw = xx - j;
        if ((unsigned)pw < LW) s += inv[(b * LW + ph) * LW + pw];
      }
    }
  }
  S[idx] = s;
}

// ---------------------------------------------------------------------------
// Stage 4: folded (bf16, NHWC):  folded[b,y,x,c] = bf16(x[b,c,y,x] * S[b,y,x])
//   LDS-staged transpose: coalesced f32 reads, contiguous 16B bf16 writes.
//   Block handles a 32(x) x 64(c) tile of one (b,y) row.
// ---------------------------------------------------------------------------
__global__ __launch_bounds__(256) void k_fold(const float* __restrict__ x,
                                              const float* __restrict__ S,
                                              __bf16* __restrict__ folded) {
  __shared__ __bf16 t[32 * CCH];                 // 4 KB, [xx][c]
  const int tid  = threadIdx.x;
  const int blk  = blockIdx.x;                   // b*512 + y*4 + xblk
  const int xblk = blk & 3;
  const int y    = (blk >> 2) & 127;
  const int b    = blk >> 9;
  const int x0   = xblk * 32;

  const int xx = tid & 31;
  const int c0 = tid >> 5;                       // 0..7
  const float sc = S[(b << 14) + (y << 7) + x0 + xx];
#pragma unroll
  for (int it = 0; it < 8; ++it) {
    int c = c0 * 8 + it;                         // wave-uniform c, xx varies
    float v = x[(((size_t)b * CCH + c) * HH + y) * WW + x0 + xx];
    t[xx * CCH + c] = (__bf16)(v * sc);
  }
  __syncthreads();
  const int ox = tid >> 3;                       // 0..31
  const int cg = (tid & 7) * 8;                  // 8-channel group
  bf16x8* dst = (bf16x8*)(folded +
      (((size_t)b * HH + y) * WW + x0 + ox) * CCH + cg);
  *dst = *(const bf16x8*)(t + ox * CCH + cg);
}

// ---------------------------------------------------------------------------
// Stage 5: weight -> bf16, reordered K:  wA[oc][g*64+ic] = w[oc][ic][i][j],
//          g = i*3 + j   (so ic is contiguous inside each K-group of 64)
// ---------------------------------------------------------------------------
__global__ __launch_bounds__(256) void k_wcvt(const float* __restrict__ w,
                                              __bf16* __restrict__ wA) {
  int idx = blockIdx.x * 256 + threadIdx.x;      // [0, 36864)
  if (idx >= CCH * KTOT) return;
  int oc = idx / KTOT;
  int k  = idx % KTOT;
  int g  = k >> 6;
  int ic = k & 63;
  int i  = g / 3, j = g % 3;
  wA[idx] = (__bf16)w[((oc * CCH + ic) * 3 + i) * 3 + j];
}

// ---------------------------------------------------------------------------
// Stage 6: implicit-GEMM conv via v_wmma_f32_16x16x32_bf16
//   M = 64 oc (4 tiles), N = pixels of one (b, oy) row (8 waves * 16),
//   K = 576 in 18 steps of 32.
//   A (weights, 64x576 bf16 = 72 KB) staged in LDS by the Tensor Data Mover
//   (one 2-D D# descriptor, wave 0 issues, TENSORcnt-tracked).
// ---------------------------------------------------------------------------
__global__ __launch_bounds__(256) void k_conv_wmma(const __bf16* __restrict__ folded,
                                                   const __bf16* __restrict__ wA,
                                                   float* __restrict__ out) {
  extern __shared__ __bf16 Alds[];               // 64*576 bf16 = 73728 B
  const int tid = threadIdx.x;

#if defined(USE_TDM)
  if (tid < 32) {                                // wave 0 only (TDM ignores EXEC)
    const unsigned long long ga = (unsigned long long)(uintptr_t)wA;
    const unsigned int lds_base = (unsigned int)(uintptr_t)(void*)Alds; // LDS offset = low 32b
    // ---- D# group 0: count=1, lds_addr, global_addr[56:0], type=2 ----
    u32x4 g0;
    g0[0] = 1u;                                          // count=1, no gather
    g0[1] = lds_base;                                    // lds_addr (bytes)
    g0[2] = (unsigned int)ga;                            // global_addr[31:0]
    g0[3] = (unsigned int)((ga >> 32) & 0x1FFFFFFull)    // global_addr[56:32]
          | (2u << 30);                                  // type = 2 ("image")
    // ---- D# group 1: 2-D tensor 64 rows x 576 cols of 2-byte elems ----
    i32x8 g1;
    g1[0] = 1 << 16;                                     // data_size=1 (2 bytes)
    g1[1] = (int)((KTOT & 0xFFFF) << 16);                // tensor_dim0 lo16 @ [63:48]
    g1[2] = (int)(((KTOT >> 16) & 0xFFFF)                // tensor_dim0 hi16 @ [79:64]
          | ((CCH & 0xFFFF) << 16));                     // tensor_dim1 lo16 @ [95:80]
    g1[3] = (int)(((CCH >> 16) & 0xFFFF)                 // tensor_dim1 hi16 @ [111:96]
          | ((KTOT & 0xFFFF) << 16));                    // tile_dim0 @ [127:112]
    g1[4] = CCH;                                         // tile_dim1 @ [143:128], tile_dim2=0
    g1[5] = KTOT;                                        // tensor_dim0_stride lo32 @ [191:160]
    g1[6] = 0;                                           // stride hi16, dim1_stride lo16
    g1[7] = 0;
    i32x4 gz4 = {0, 0, 0, 0};                            // groups 2/3 unused (<=2D)
    i32x8 gz8 = {0, 0, 0, 0, 0, 0, 0, 0};                // trailing group (unused)
    __builtin_amdgcn_tensor_load_to_lds(g0, g1, gz4, gz4, gz8, 0);
    __builtin_amdgcn_s_wait_tensorcnt(0);
  }
#else
  {
    const float4* src = (const float4*)wA;
    float4*       dst = (float4*)Alds;
#pragma unroll
    for (int i = tid; i < (CCH * KTOT * 2) / 16; i += 256) dst[i] = src[i];
  }
#endif
  __syncthreads();

  const int row  = blockIdx.x;                   // b*126 + oy
  const int b    = row / LW;
  const int oy   = row % LW;
  const int wave = tid >> 5;
  const int lane = tid & 31;
  const int nl   = lane & 15;                    // N (pixel) / M (oc) within tile
  const int hi   = lane >> 4;                    // half-wave select
  const int ox0  = wave * 16;                    // pixel base of this wave

  v8f acc0 = {}, acc1 = {}, acc2 = {}, acc3 = {};

#pragma unroll
  for (int kk = 0; kk < KTOT; kk += 32) {
    const int g  = kk >> 6;                      // (i,j) group, compile-time
    const int di = g / 3, dj = g % 3;
    // ---- B fragment: lane holds two 8-element ic runs (NHWC layout) ----
    const int icb = (kk & 63) + hi * 8;
    const __bf16* bp = folded
        + (((size_t)b * HH + (oy + di)) * WW + (ox0 + nl + dj)) * CCH + icb;
    __builtin_prefetch(bp + CCH, 0, 1);          // global_prefetch_b8 flavor
    Frag16 Bf;
    Bf.h[0] = *(const bf16x8*)bp;
    Bf.h[1] = *(const bf16x8*)(bp + 16);

    // ---- A fragments from LDS: lane = row oc, two 8-element runs ----
    const int abase = kk + hi * 8;
    const __bf16* ap0 = Alds + (size_t)(0 * 16 + nl) * KTOT + abase;
    const __bf16* ap1 = Alds + (size_t)(1 * 16 + nl) * KTOT + abase;
    const __bf16* ap2 = Alds + (size_t)(2 * 16 + nl) * KTOT + abase;
    const __bf16* ap3 = Alds + (size_t)(3 * 16 + nl) * KTOT + abase;
    Frag16 A0, A1, A2, A3;
    A0.h[0] = *(const bf16x8*)ap0;  A0.h[1] = *(const bf16x8*)(ap0 + 16);
    A1.h[0] = *(const bf16x8*)ap1;  A1.h[1] = *(const bf16x8*)(ap1 + 16);
    A2.h[0] = *(const bf16x8*)ap2;  A2.h[1] = *(const bf16x8*)(ap2 + 16);
    A3.h[0] = *(const bf16x8*)ap3;  A3.h[1] = *(const bf16x8*)(ap3 + 16);

    acc0 = __builtin_amdgcn_wmma_f32_16x16x32_bf16(false, A0.v, false, Bf.v,
                                                   (short)0, acc0, false, false);
    acc1 = __builtin_amdgcn_wmma_f32_16x16x32_bf16(false, A1.v, false, Bf.v,
                                                   (short)0, acc1, false, false);
    acc2 = __builtin_amdgcn_wmma_f32_16x16x32_bf16(false, A2.v, false, Bf.v,
                                                   (short)0, acc2, false, false);
    acc3 = __builtin_amdgcn_wmma_f32_16x16x32_bf16(false, A3.v, false, Bf.v,
                                                   (short)0, acc3, false, false);
  }

  // ---- epilogue: D layout VGPR r -> oc = mtile*16 + r + 8*hi, col = nl ----
  const int ox = ox0 + nl;
  if (ox < LW) {
    float* op = out + ((size_t)b * CCH * LW + oy) * LW + ox;  // oc stride LW*LW
#pragma unroll
    for (int r = 0; r < 8; ++r) {
      int oc = r + hi * 8;
      op[(size_t)(oc +  0) * LW * LW] = acc0[r];
      op[(size_t)(oc + 16) * LW * LW] = acc1[r];
      op[(size_t)(oc + 32) * LW * LW] = acc2[r];
      op[(size_t)(oc + 48) * LW * LW] = acc3[r];
    }
  }
}

// ---------------------------------------------------------------------------
// Launch
// ---------------------------------------------------------------------------
extern "C" void kernel_launch(void* const* d_in, const int* in_sizes, int n_in,
                              void* d_out, int out_size, void* d_ws, size_t ws_size,
                              hipStream_t stream) {
  const float* x = (const float*)d_in[0];   // [8,64,128,128]
  const float* w = (const float*)d_in[1];   // [64,64,3,3] (pre-normalized)
  float* out = (float*)d_out;               // [8,64,126,126]

  char* ws = (char*)d_ws;
  float*  sq     = (float*)(ws + 0);                // 512 KB
  float*  inv    = (float*)(ws + (size_t)524288);   // ~496 KB used
  float*  S      = (float*)(ws + (size_t)1048576);  // 512 KB
  __bf16* wA     = (__bf16*)(ws + (size_t)1572864); // 72 KB
  __bf16* folded = (__bf16*)(ws + (size_t)1703936); // 16 MB (+guard)

  k_sq   <<<512,   256, 0, stream>>>(x, sq);
  k_inv  <<<(BN * LW * LW + 255) / 256, 256, 0, stream>>>(sq, inv);
  k_scale<<<512,   256, 0, stream>>>(inv, S);
  k_fold <<<BN * HH * 4, 256, 0, stream>>>(x, S, folded);
  k_wcvt <<<(CCH * KTOT + 255) / 256, 256, 0, stream>>>(w, wA);
  k_conv_wmma<<<BN * LW, 256, CCH * KTOT * sizeof(__bf16), stream>>>(folded, wA, out);
}
